// LongformerSelfAttentionPegasus_40252433498488
// MI455X (gfx1250) — matrix operand order repaired and checked by reading, not compiled, not run on hardware
//
#include <hip/hip_runtime.h>

#define B_  2
#define S_  4096
#define D_  768
#define H_  12
#define HD_ 64
#define W_  256
#define G_  128
#define NB_ 16

typedef __bf16 bf16_t;
typedef __bf16 v16bf __attribute__((ext_vector_type(16)));
typedef __bf16 v8bf  __attribute__((ext_vector_type(8)));
typedef float  v8f   __attribute__((ext_vector_type(8)));
typedef unsigned int v4u __attribute__((ext_vector_type(4)));
typedef int          v8i __attribute__((ext_vector_type(8)));
typedef int          v4i __attribute__((ext_vector_type(4)));

__device__ __forceinline__ bf16_t f2bf(float f) {
  unsigned u = __float_as_uint(f);
  unsigned r = u + 0x7fffu + ((u >> 16) & 1u);   // round-to-nearest-even
  unsigned short hx = (unsigned short)(r >> 16);
  bf16_t b;
  __builtin_memcpy(&b, &hx, 2);
  return b;
}

__device__ __forceinline__ v8f v8f_zero() {
  v8f z;
#pragma unroll
  for (int i = 0; i < 8; ++i) z[i] = 0.0f;
  return z;
}

__device__ __forceinline__ v8f wmma_bf16(v16bf a, v16bf b, v8f c) {
  return __builtin_amdgcn_wmma_f32_16x16x32_bf16(false, a, false, b, (short)0, c, false, false);
}

// B-fragment: this lane's 16 contiguous K values (32 bytes).
__device__ __forceinline__ v16bf frag16(const bf16_t* p) {
  union { v16bf v; struct { v8bf lo, hi; } h; } u;
  u.h.lo = *(const v8bf*)p;
  u.h.hi = *(const v8bf*)(p + 8);
  return u.v;
}

// A-fragment (16x32 bf16): rowp points at this lane's row, 32 consecutive K.
// lane<16: K {0..15}; lane>=16: K {16..31}.
__device__ __forceinline__ v16bf fragA(const bf16_t* rowp, int ln) {
  const int k0 = (ln & 16) ? 16 : 0;
  union { v16bf v; struct { v8bf lo, hi; } h; } u;
  u.h.lo = *(const v8bf*)(rowp + k0);
  u.h.hi = *(const v8bf*)(rowp + k0 + 8);
  return u.v;
}

// CDNA5 async memory->LDS copy (ASYNCcnt-tracked, no VGPR round trip).
__device__ __forceinline__ void async_load_b128(void* lds_dst, const void* gsrc) {
  unsigned l = (unsigned)(size_t)lds_dst;               // low 32 bits = LDS offset
  unsigned long long g = (unsigned long long)gsrc;
  asm volatile("global_load_async_to_lds_b128 %0, %1, off"
               :: "v"(l), "v"(g) : "memory");
}
__device__ __forceinline__ void wait_async() {
  asm volatile("s_wait_asynccnt 0x0" ::: "memory");
}

// CDNA5 Tensor Data Mover: 2-D tile load global->LDS (bf16 elements).
// Descriptor packing per CDNA5 ISA (data_size=1 -> 2B).
// 6-arg builtin form (amdgpu-toolchain clang-23).
__device__ __forceinline__ void tdm_load_2d_bf16(unsigned lds_off, const void* gaddr,
                                                 unsigned tensor_d0, unsigned tensor_d1,
                                                 unsigned tile_d0, unsigned tile_d1,
                                                 unsigned stride0_elems)
{
  unsigned long long ga = (unsigned long long)gaddr;
  v4u g0;
  g0[0] = 1u;                                            // count=1, user descriptor
  g0[1] = lds_off;                                       // lds_addr (bytes)
  g0[2] = (unsigned)(ga & 0xffffffffu);                  // global_addr[31:0]
  g0[3] = (unsigned)((ga >> 32) & 0x01ffffffu) | (2u << 30); // addr[56:32] | type=2
  v8i g1;
  g1[0] = (int)(1u << 16);                               // data_size=1 (2 bytes)
  g1[1] = (int)((tensor_d0 & 0xffffu) << 16);            // tensor_dim0[15:0] @ bit48
  g1[2] = (int)((tensor_d0 >> 16) | ((tensor_d1 & 0xffffu) << 16));
  g1[3] = (int)((tensor_d1 >> 16) | (tile_d0 << 16));    // tile_dim0 @ bit112
  g1[4] = (int)(tile_d1 & 0xffffu);                      // tile_dim1, tile_dim2=0
  g1[5] = (int)stride0_elems;                            // tensor_dim0_stride[31:0]
  g1[6] = 0;
  g1[7] = 0;
  v4i z4; z4[0] = 0; z4[1] = 0; z4[2] = 0; z4[3] = 0;
  v8i z8;
#pragma unroll
  for (int i = 0; i < 8; ++i) z8[i] = 0;
  __builtin_amdgcn_tensor_load_to_lds(g0, g1, z4, z4, z8, 0);
}

// ---------------------------------------------------------------- convert
__global__ void __launch_bounds__(256)
cvt_f32_bf16(const float* __restrict__ src, bf16_t* __restrict__ dst, int n) {
  int i = blockIdx.x * 256 + threadIdx.x;
  if (i < n) dst[i] = f2bf(src[i]);
}

// ---------------------------------------------------------------- GEMM
// C[M=8192, N=768] = A[M,768] * W[768,N]; tile 128x128, 8 waves of 64x32.
// A tile staged via Tensor Data Mover; W tile transposed via LDS scatter.
// mode 0: out_bf16[b,h,s,hd] = bf16((acc + bias) * scale)
// mode 1: yout[row,f] = acc + bias + resid[row,f]   (fp32)
__global__ void __launch_bounds__(256)
gemm_proj(const bf16_t* __restrict__ A, const bf16_t* __restrict__ Wm,
          const float* __restrict__ bias, float scale,
          bf16_t* __restrict__ outb, const float* __restrict__ resid,
          float* __restrict__ yout, int mode)
{
  __shared__ __align__(16) bf16_t sA [128][32];   // [row][k]
  __shared__ __align__(16) bf16_t sWT[128][32];   // [feature][k] (transposed)

  const int tid = threadIdx.x;
  const int wv  = tid >> 5;
  const int ln  = tid & 31;
  const int wm  = wv >> 2;                 // 0..1 -> 64 rows
  const int wn  = wv & 3;                  // 0..3 -> 32 cols
  const int m0  = blockIdx.y * 128;
  const int n0  = blockIdx.x * 128;
  const int rhalf = (ln & 16) ? 8 : 0;

  v8f acc[4][2];
#pragma unroll
  for (int mt = 0; mt < 4; ++mt)
#pragma unroll
    for (int nt = 0; nt < 2; ++nt) acc[mt][nt] = v8f_zero();

  for (int kb = 0; kb < D_; kb += 32) {
    __syncthreads();
    if (wv == 0) {
      // TDM: 128-row x 32-col bf16 tile out of the 8192x768 activation tensor
      tdm_load_2d_bf16((unsigned)(size_t)(void*)&sA[0][0],
                       A + (size_t)m0 * D_ + kb,
                       (unsigned)D_, (unsigned)(B_ * S_),
                       32u, 128u, (unsigned)D_);
      __builtin_amdgcn_s_wait_tensorcnt(0);
    }
    {
      int kk = tid & 31, fg = tid >> 5;
      const bf16_t* src = Wm + (size_t)(kb + kk) * D_ + n0 + fg * 16;
      __align__(16) bf16_t tmp[16];
      *(v8bf*)tmp       = *(const v8bf*)src;
      *(v8bf*)(tmp + 8) = *(const v8bf*)(src + 8);
      if (kb + 32 < D_) __builtin_prefetch(src + 32 * D_, 0, 1);
#pragma unroll
      for (int i = 0; i < 16; ++i) sWT[fg * 16 + i][kk] = tmp[i];
    }
    __syncthreads();

    v16bf wfr[2];
#pragma unroll
    for (int nt = 0; nt < 2; ++nt)
      wfr[nt] = frag16(&sWT[wn * 32 + nt * 16 + (ln & 15)][(ln & 16) ? 16 : 0]);
#pragma unroll
    for (int mt = 0; mt < 4; ++mt) {
      v16bf af = fragA(&sA[wm * 64 + mt * 16 + (ln & 15)][0], ln);
#pragma unroll
      for (int nt = 0; nt < 2; ++nt)
        acc[mt][nt] = wmma_bf16(af, wfr[nt], acc[mt][nt]);
    }
  }

#pragma unroll
  for (int mt = 0; mt < 4; ++mt)
#pragma unroll
    for (int nt = 0; nt < 2; ++nt)
#pragma unroll
      for (int r = 0; r < 8; ++r) {
        int row = m0 + wm * 64 + mt * 16 + rhalf + r;
        int f   = n0 + wn * 32 + nt * 16 + (ln & 15);
        float val = acc[mt][nt][r] + bias[f];
        if (mode == 0) {
          val *= scale;
          int bb = row >> 12;            // / S_
          int ss = row & (S_ - 1);
          int hh = f >> 6;
          int hd = f & 63;
          outb[(((size_t)bb * H_ + hh) * S_ + ss) * HD_ + hd] = f2bf(val);
        } else {
          yout[(size_t)row * D_ + f] = val + resid[(size_t)row * D_ + f];
        }
      }
}

// ---------------------------------------------------------------- band attention
// one block per (b, h, n); 8 waves x 32 query rows; flash softmax over
// 28 chunks of 32 key-columns: cols [0,128) global keys, [128,896) band.
__global__ void __launch_bounds__(256)
band_attn(const bf16_t* __restrict__ q, const bf16_t* __restrict__ k,
          const bf16_t* __restrict__ v, bf16_t* __restrict__ ctx)
{
  __shared__ __align__(16) bf16_t sK [32][64];       // [key][dim]
  __shared__ __align__(16) bf16_t sVT[64][32];       // [dim][key]
  __shared__ __align__(16) bf16_t sP [8][32][32];    // per-wave probs [row][key]

  const int tid = threadIdx.x;
  const int wv  = tid >> 5;
  const int ln  = tid & 31;
  const int rhalf = (ln & 16) ? 8 : 0;

  int idx = blockIdx.x;
  const int n = idx % NB_;
  const int h = (idx / NB_) % H_;
  const int b = idx / (NB_ * H_);

  const size_t hoff = ((size_t)(b * H_ + h)) * S_ * HD_;
  const bf16_t* qh = q + hoff;
  const bf16_t* kh = k + hoff;
  const bf16_t* vh = v + hoff;

  const int qbase = n * W_ + wv * 32;

  v16bf qf[2][2];
#pragma unroll
  for (int mt = 0; mt < 2; ++mt)
#pragma unroll
    for (int ks = 0; ks < 2; ++ks) {
      int row = qbase + mt * 16 + (ln & 15);
      qf[mt][ks] = fragA(qh + (size_t)row * HD_ + ks * 32, ln);
    }

  v8f O[2][4];
  float mr[2][8], lr[2][8];
#pragma unroll
  for (int mt = 0; mt < 2; ++mt) {
#pragma unroll
    for (int dt = 0; dt < 4; ++dt) O[mt][dt] = v8f_zero();
#pragma unroll
    for (int r = 0; r < 8; ++r) { mr[mt][r] = -1e30f; lr[mt][r] = 0.0f; }
  }

  for (int c = 0; c < 28; ++c) {
    __syncthreads();
    {   // K row staged async to LDS; V row transposed through VGPRs
      int key = tid >> 3;            // 0..31
      int seg = tid & 7;             // 8 dims
      int col = c * 32 + key;
      int src;
      if (col < G_) src = col;
      else {
        int kpos = n * W_ + (col - G_) - W_;
        src = kpos < 0 ? 0 : (kpos >= S_ ? S_ - 1 : kpos);
      }
      async_load_b128(&sK[key][seg * 8], kh + (size_t)src * HD_ + seg * 8);
      __align__(16) bf16_t tmp[8];
      *(v8bf*)tmp = *(const v8bf*)(vh + (size_t)src * HD_ + seg * 8);
#pragma unroll
      for (int i = 0; i < 8; ++i) sVT[seg * 8 + i][key] = tmp[i];
    }
    wait_async();
    __syncthreads();

    v16bf kf[2][2];
#pragma unroll
    for (int nt = 0; nt < 2; ++nt)
#pragma unroll
      for (int ks = 0; ks < 2; ++ks)
        kf[nt][ks] = frag16(&sK[nt * 16 + (ln & 15)][ks * 32 + ((ln & 16) ? 16 : 0)]);

    v8f sc[2][2];
#pragma unroll
    for (int mt = 0; mt < 2; ++mt)
#pragma unroll
      for (int nt = 0; nt < 2; ++nt) {
        v8f s = v8f_zero();
        s = wmma_bf16(qf[mt][0], kf[nt][0], s);
        s = wmma_bf16(qf[mt][1], kf[nt][1], s);
        sc[mt][nt] = s;
      }

#pragma unroll
    for (int mt = 0; mt < 2; ++mt) {
      float tmax[8];
#pragma unroll
      for (int r = 0; r < 8; ++r) {
        int qi = wv * 32 + mt * 16 + rhalf + r;          // row within block
#pragma unroll
        for (int nt = 0; nt < 2; ++nt) {
          int col = c * 32 + nt * 16 + (ln & 15);
          float sval = sc[mt][nt][r];
          if (col >= G_) {
            int bcol = col - G_;
            int kpos = n * W_ + bcol - W_;
            int rel  = bcol - W_ - qi;
            bool ok = (rel <= W_) && (rel >= -W_) && (kpos >= G_) && (kpos < S_);
            sval = ok ? sval : -1e9f;
          }
          sc[mt][nt][r] = sval;
        }
        tmax[r] = fmaxf(sc[mt][0][r], sc[mt][1][r]);
      }
#pragma unroll
      for (int r = 0; r < 8; ++r) {
        float t = tmax[r];
        t = fmaxf(t, __shfl_xor(t, 1));
        t = fmaxf(t, __shfl_xor(t, 2));
        t = fmaxf(t, __shfl_xor(t, 4));
        t = fmaxf(t, __shfl_xor(t, 8));
        tmax[r] = t;
      }
      float psum[8];
#pragma unroll
      for (int r = 0; r < 8; ++r) {
        float mnew = fmaxf(mr[mt][r], tmax[r]);
        float corr = __expf(mr[mt][r] - mnew);
        mr[mt][r] = mnew;
        lr[mt][r] *= corr;
#pragma unroll
        for (int dt = 0; dt < 4; ++dt) O[mt][dt][r] = O[mt][dt][r] * corr;
        float p0 = __expf(sc[mt][0][r] - mnew);
        float p1 = __expf(sc[mt][1][r] - mnew);
        int prow = mt * 16 + rhalf + r;
        sP[wv][prow][ln & 15]        = f2bf(p0);
        sP[wv][prow][16 + (ln & 15)] = f2bf(p1);
        psum[r] = p0 + p1;
      }
#pragma unroll
      for (int r = 0; r < 8; ++r) {
        float t = psum[r];
        t += __shfl_xor(t, 1);
        t += __shfl_xor(t, 2);
        t += __shfl_xor(t, 4);
        t += __shfl_xor(t, 8);
        lr[mt][r] += t;
      }
    }

    v16bf vf[4];
#pragma unroll
    for (int dt = 0; dt < 4; ++dt)
      vf[dt] = frag16(&sVT[dt * 16 + (ln & 15)][(ln & 16) ? 16 : 0]);
#pragma unroll
    for (int mt = 0; mt < 2; ++mt) {
      v16bf pf = fragA(&sP[wv][mt * 16 + (ln & 15)][0], ln);
#pragma unroll
      for (int dt = 0; dt < 4; ++dt)
        O[mt][dt] = wmma_bf16(pf, vf[dt], O[mt][dt]);
    }
  }

#pragma unroll
  for (int mt = 0; mt < 2; ++mt)
#pragma unroll
    for (int r = 0; r < 8; ++r) {
      float inv = 1.0f / lr[mt][r];
      int srow = qbase + mt * 16 + rhalf + r;
#pragma unroll
      for (int dt = 0; dt < 4; ++dt) {
        int dim = dt * 16 + (ln & 15);
        ctx[((size_t)b * S_ + srow) * D_ + h * HD_ + dim] = f2bf(O[mt][dt][r] * inv);
      }
    }
}

// ---------------------------------------------------------------- global attention
// rows [0,128) attend to all 4096 keys; one block per (b,h); wave = 16 rows.
__global__ void __launch_bounds__(256)
glob_attn(const bf16_t* __restrict__ qg, const bf16_t* __restrict__ kg,
          const bf16_t* __restrict__ vg, bf16_t* __restrict__ ctx)
{
  __shared__ __align__(16) bf16_t sK [32][64];
  __shared__ __align__(16) bf16_t sVT[64][32];
  __shared__ __align__(16) bf16_t sP [8][16][32];

  const int tid = threadIdx.x;
  const int wv  = tid >> 5;
  const int ln  = tid & 31;
  const int rhalf = (ln & 16) ? 8 : 0;

  const int h = blockIdx.x % H_;
  const int b = blockIdx.x / H_;
  const size_t hoff = ((size_t)(b * H_ + h)) * S_ * HD_;
  const bf16_t* qh = qg + hoff;
  const bf16_t* kh = kg + hoff;
  const bf16_t* vh = vg + hoff;

  const int qrow0 = wv * 16;
  v16bf qf[2];
#pragma unroll
  for (int ks = 0; ks < 2; ++ks) {
    int row = qrow0 + (ln & 15);
    qf[ks] = fragA(qh + (size_t)row * HD_ + ks * 32, ln);
  }

  v8f O[4];
  float mr[8], lr[8];
#pragma unroll
  for (int dt = 0; dt < 4; ++dt) O[dt] = v8f_zero();
#pragma unroll
  for (int r = 0; r < 8; ++r) { mr[r] = -1e30f; lr[r] = 0.0f; }

  for (int c = 0; c < S_ / 32; ++c) {
    __syncthreads();
    {
      int key = tid >> 3;
      int seg = tid & 7;
      int src = c * 32 + key;
      async_load_b128(&sK[key][seg * 8], kh + (size_t)src * HD_ + seg * 8);
      __align__(16) bf16_t tmp[8];
      *(v8bf*)tmp = *(const v8bf*)(vh + (size_t)src * HD_ + seg * 8);
#pragma unroll
      for (int i = 0; i < 8; ++i) sVT[seg * 8 + i][key] = tmp[i];
    }
    wait_async();
    __syncthreads();

    v8f sc[2];
#pragma unroll
    for (int nt = 0; nt < 2; ++nt) {
      v16bf k0 = frag16(&sK[nt * 16 + (ln & 15)][((ln & 16) ? 16 : 0)]);
      v16bf k1 = frag16(&sK[nt * 16 + (ln & 15)][32 + ((ln & 16) ? 16 : 0)]);
      v8f s = v8f_zero();
      s = wmma_bf16(qf[0], k0, s);
      s = wmma_bf16(qf[1], k1, s);
      sc[nt] = s;
    }

    float tmax[8];
#pragma unroll
    for (int r = 0; r < 8; ++r) {
      float t = fmaxf(sc[0][r], sc[1][r]);
      t = fmaxf(t, __shfl_xor(t, 1));
      t = fmaxf(t, __shfl_xor(t, 2));
      t = fmaxf(t, __shfl_xor(t, 4));
      t = fmaxf(t, __shfl_xor(t, 8));
      tmax[r] = t;
    }
    float psum[8];
#pragma unroll
    for (int r = 0; r < 8; ++r) {
      float mnew = fmaxf(mr[r], tmax[r]);
      float corr = __expf(mr[r] - mnew);
      mr[r] = mnew;
      lr[r] *= corr;
#pragma unroll
      for (int dt = 0; dt < 4; ++dt) O[dt][r] = O[dt][r] * corr;
      float p0 = __expf(sc[0][r] - mnew);
      float p1 = __expf(sc[1][r] - mnew);
      int prow = rhalf + r;
      sP[wv][prow][ln & 15]        = f2bf(p0);
      sP[wv][prow][16 + (ln & 15)] = f2bf(p1);
      psum[r] = p0 + p1;
    }
#pragma unroll
    for (int r = 0; r < 8; ++r) {
      float t = psum[r];
      t += __shfl_xor(t, 1);
      t += __shfl_xor(t, 2);
      t += __shfl_xor(t, 4);
      t += __shfl_xor(t, 8);
      lr[r] += t;
    }

    v16bf pf = fragA(&sP[wv][(ln & 15)][0], ln);
#pragma unroll
    for (int dt = 0; dt < 4; ++dt) {
      v16bf vf = frag16(&sVT[dt * 16 + (ln & 15)][(ln & 16) ? 16 : 0]);
      O[dt] = wmma_bf16(pf, vf, O[dt]);
    }
  }

#pragma unroll
  for (int r = 0; r < 8; ++r) {
    float inv = 1.0f / lr[r];
    int srow = qrow0 + rhalf + r;          // < 128
#pragma unroll
    for (int dt = 0; dt < 4; ++dt) {
      int dim = dt * 16 + (ln & 15);
      ctx[((size_t)b * S_ + srow) * D_ + h * HD_ + dim] = f2bf(O[dt][r] * inv);
    }
  }
}

// ---------------------------------------------------------------- layernorm
__global__ void __launch_bounds__(256)
layernorm(const float* __restrict__ y, const float* __restrict__ gamma,
          const float* __restrict__ beta, float* __restrict__ out)
{
  __shared__ float red[2][8];
  const int row = blockIdx.x;
  const float* yr = y + (size_t)row * D_;
  float s = 0.0f, s2 = 0.0f;
  for (int i = threadIdx.x; i < D_; i += 256) {
    float t = yr[i];
    s += t; s2 += t * t;
  }
#pragma unroll
  for (int m = 1; m < 32; m <<= 1) { s += __shfl_xor(s, m); s2 += __shfl_xor(s2, m); }
  if ((threadIdx.x & 31) == 0) { red[0][threadIdx.x >> 5] = s; red[1][threadIdx.x >> 5] = s2; }
  __syncthreads();
  float ts = 0.0f, ts2 = 0.0f;
#pragma unroll
  for (int i = 0; i < 8; ++i) { ts += red[0][i]; ts2 += red[1][i]; }
  float mu  = ts / (float)D_;
  float var = ts2 / (float)D_ - mu * mu;
  float rs  = rsqrtf(var + 1e-5f);
  for (int i = threadIdx.x; i < D_; i += 256)
    out[(size_t)row * D_ + i] = (yr[i] - mu) * rs * gamma[i] + beta[i];
}

// ---------------------------------------------------------------- launch
extern "C" void kernel_launch(void* const* d_in, const int* in_sizes, int n_in,
                              void* d_out, int out_size, void* d_ws, size_t ws_size,
                              hipStream_t stream)
{
  (void)in_sizes; (void)n_in; (void)out_size; (void)ws_size;
  const float* hs    = (const float*)d_in[0];
  const float* Wq    = (const float*)d_in[3];
  const float* bq    = (const float*)d_in[4];
  const float* Wk    = (const float*)d_in[5];
  const float* bk    = (const float*)d_in[6];
  const float* Wv    = (const float*)d_in[7];
  const float* bv    = (const float*)d_in[8];
  const float* Wqg   = (const float*)d_in[9];
  const float* bqg   = (const float*)d_in[10];
  const float* Wkg   = (const float*)d_in[11];
  const float* bkg   = (const float*)d_in[12];
  const float* Wvg   = (const float*)d_in[13];
  const float* bvg   = (const float*)d_in[14];
  const float* Wo    = (const float*)d_in[15];
  const float* bo    = (const float*)d_in[16];
  const float* gamma = (const float*)d_in[17];
  const float* beta  = (const float*)d_in[18];

  const size_t BSD = (size_t)B_ * S_ * D_;       // 6,291,456
  const size_t DD  = (size_t)D_ * D_;            // 589,824

  char* p = (char*)d_ws;
  auto take = [&](size_t bytes) -> void* {
    void* r = (void*)p;
    p += (bytes + 255) & ~(size_t)255;
    return r;
  };
  bf16_t* hsb = (bf16_t*)take(BSD * 2);
  bf16_t* wb[7];
  for (int i = 0; i < 7; ++i) wb[i] = (bf16_t*)take(DD * 2);
  bf16_t* qa  = (bf16_t*)take(BSD * 2);
  bf16_t* ka  = (bf16_t*)take(BSD * 2);
  bf16_t* va  = (bf16_t*)take(BSD * 2);
  bf16_t* qga = (bf16_t*)take(BSD * 2);
  bf16_t* kga = (bf16_t*)take(BSD * 2);
  bf16_t* vga = (bf16_t*)take(BSD * 2);
  bf16_t* ctx = (bf16_t*)take(BSD * 2);
  float*  yb  = (float*)take(BSD * 4);

  cvt_f32_bf16<<<(int)((BSD + 255) / 256), 256, 0, stream>>>(hs, hsb, (int)BSD);
  const float* wsrc[7] = {Wq, Wk, Wv, Wqg, Wkg, Wvg, Wo};
  for (int i = 0; i < 7; ++i)
    cvt_f32_bf16<<<(int)((DD + 255) / 256), 256, 0, stream>>>(wsrc[i], wb[i], (int)DD);

  dim3 gg(D_ / 128, (B_ * S_) / 128);            // (6, 64)
  gemm_proj<<<gg, 256, 0, stream>>>(hsb, wb[0], bq,  0.125f, qa,  nullptr, nullptr, 0);
  gemm_proj<<<gg, 256, 0, stream>>>(hsb, wb[1], bk,  1.0f,   ka,  nullptr, nullptr, 0);
  gemm_proj<<<gg, 256, 0, stream>>>(hsb, wb[2], bv,  1.0f,   va,  nullptr, nullptr, 0);
  gemm_proj<<<gg, 256, 0, stream>>>(hsb, wb[3], bqg, 0.125f, qga, nullptr, nullptr, 0);
  gemm_proj<<<gg, 256, 0, stream>>>(hsb, wb[4], bkg, 1.0f,   kga, nullptr, nullptr, 0);
  gemm_proj<<<gg, 256, 0, stream>>>(hsb, wb[5], bvg, 1.0f,   vga, nullptr, nullptr, 0);

  band_attn<<<B_ * H_ * NB_, 256, 0, stream>>>(qa, ka, va, ctx);
  glob_attn<<<B_ * H_, 256, 0, stream>>>(qga, kga, vga, ctx);

  gemm_proj<<<gg, 256, 0, stream>>>(ctx, wb[6], bo, 1.0f, nullptr, hs, yb, 1);
  layernorm<<<B_ * S_, 256, 0, stream>>>(yb, gamma, beta, (float*)d_out);
}
